// CPCCriterion_16355235463283
// MI455X (gfx1250) — compile-verified, matrix operand order checked by reading, not run on hardware
//
#include <hip/hip_runtime.h>
#include <hip/hip_bf16.h>
#include <stddef.h>

// ---------------------------------------------------------------------------
// CPC InfoNCE criterion for MI455X (gfx1250, wave32, WMMA).
//   B=8, T=256, C_ctx=512, C_enc=256, P=12, K=128, TEMP=0.07
// GEMMs run on the matrix pipe via v_wmma_f32_16x16x32_bf16 with a software-
// pipelined (double-buffered) fragment stream so loads overlap the WMMAs.
// The gather-bound logit phase streams the L2-resident normalized pool.
// ---------------------------------------------------------------------------

#define CPC_B     8
#define CPC_T     256
#define CPC_CCTX  512
#define CPC_CENC  256
#define CPC_P     12
#define CPC_K     128
#define CPC_ROWS  (CPC_B * CPC_T)         // 2048
#define CPC_NEGROWS (CPC_B * (CPC_T - 1)) // 2040 (neg_idx second dim)
#define CPC_INVT  (1.0f / 0.07f)

typedef __attribute__((ext_vector_type(16))) __bf16        v16bf;
typedef __attribute__((ext_vector_type(8)))  float         v8f;
typedef __attribute__((ext_vector_type(8)))  unsigned int  v8u;

static __device__ __forceinline__ v16bf as_v16bf(v8u u) {
    union { v8u u; v16bf b; } x; x.u = u; return x.b;
}

// float -> bf16, round-to-nearest-even
static __device__ __forceinline__ unsigned short f2bf_rne(float f) {
    unsigned int u = __float_as_uint(f);
    u += 0x7FFFu + ((u >> 16) & 1u);
    return (unsigned short)(u >> 16);
}

// ------------------------------ conversion ---------------------------------
__global__ void cpc_f32_to_bf16_kernel(const float* __restrict__ src,
                                       unsigned short* __restrict__ dst, int n) {
    int i = blockIdx.x * blockDim.x + threadIdx.x;
    if (i < n) dst[i] = f2bf_rne(src[i]);
}

// ------------------------- pool row normalization --------------------------
// One wave (32 lanes) per 256-float row; 8 waves per 256-thread block.
__global__ void cpc_pool_norm_kernel(const float* __restrict__ enc,
                                     float* __restrict__ pool) {
    const int wave = threadIdx.x >> 5;
    const int lane = threadIdx.x & 31;
    const int row  = (blockIdx.x << 3) + wave;           // 2048 rows total
    const float4* s = (const float4*)(enc + (size_t)row * CPC_CENC);
    float4 v0 = s[lane];
    float4 v1 = s[lane + 32];
    float ss = v0.x * v0.x + v0.y * v0.y + v0.z * v0.z + v0.w * v0.w
             + v1.x * v1.x + v1.y * v1.y + v1.z * v1.z + v1.w * v1.w;
    #pragma unroll
    for (int off = 16; off > 0; off >>= 1) ss += __shfl_xor(ss, off, 32);
    const float inv = rsqrtf(ss);
    float4* d = (float4*)(pool + (size_t)row * CPC_CENC);
    d[lane]      = make_float4(v0.x * inv, v0.y * inv, v0.z * inv, v0.w * inv);
    d[lane + 32] = make_float4(v1.x * inv, v1.y * inv, v1.z * inv, v1.w * inv);
}

// ------------------------------- WMMA GEMM ---------------------------------
// z(2048 x 256) = ctx_bf(2048 x 512) @ Wk_bf(256 x 512)^T   (f32 accumulate)
// One wave per 16(M) x 64(N) tile: 4 accumulators reuse the A fragment.
// A frag: lane l (l<16): row M=l&15, K = 0..15 ; lanes 16..31: same M, K=16..31.
// B frag: lane l:        col N=l&15, K split identically (W rows are contiguous
//         in the K=C_ctx dimension, so each lane does one 32-byte load).
// C/D   : VGPR r holds M=r (lanes 0-15) / M=r+8 (lanes 16-31), N = lane&15.
// Fragments are double-buffered: step kk's WMMAs consume registers loaded in
// step kk-32, so the 10 in-flight b128 loads hide latency behind the WMMAs.
__global__ void __launch_bounds__(32)
cpc_gemm_zhat_kernel(const unsigned short* __restrict__ Abf,
                     const unsigned short* __restrict__ Wbf,
                     float* __restrict__ D) {
    const int lane = threadIdx.x;              // 0..31, full wave (EXEC all 1s)
    const int m0   = blockIdx.y << 4;          // M tile base (2048/16 = 128)
    const int n0   = blockIdx.x << 6;          // N tile base (256/64 = 4)
    const int lr   = lane & 15;
    const int kb   = (lane >> 4) << 4;         // 0 or 16

    const unsigned short* arow = Abf + (size_t)(m0 + lr) * CPC_CCTX + kb;
    const unsigned short* b0   = Wbf + (size_t)(n0 +  0 + lr) * CPC_CCTX + kb;
    const unsigned short* b1   = Wbf + (size_t)(n0 + 16 + lr) * CPC_CCTX + kb;
    const unsigned short* b2   = Wbf + (size_t)(n0 + 32 + lr) * CPC_CCTX + kb;
    const unsigned short* b3   = Wbf + (size_t)(n0 + 48 + lr) * CPC_CCTX + kb;

    v8f acc0 = {}; v8f acc1 = {}; v8f acc2 = {}; v8f acc3 = {};

    // prologue: fragment set for kk = 0
    v8u a_c  = *(const v8u*)(arow);
    v8u w0_c = *(const v8u*)(b0);
    v8u w1_c = *(const v8u*)(b1);
    v8u w2_c = *(const v8u*)(b2);
    v8u w3_c = *(const v8u*)(b3);

    #pragma unroll
    for (int kk = 0; kk < CPC_CCTX - 32; kk += 32) {
        const int kn = kk + 32;
        // issue next-step loads before consuming the current fragments
        v8u a_n  = *(const v8u*)(arow + kn);
        v8u w0_n = *(const v8u*)(b0 + kn);
        v8u w1_n = *(const v8u*)(b1 + kn);
        v8u w2_n = *(const v8u*)(b2 + kn);
        v8u w3_n = *(const v8u*)(b3 + kn);

        acc0 = __builtin_amdgcn_wmma_f32_16x16x32_bf16(false, as_v16bf(a_c), false,
                  as_v16bf(w0_c), (short)0, acc0, false, false);
        acc1 = __builtin_amdgcn_wmma_f32_16x16x32_bf16(false, as_v16bf(a_c), false,
                  as_v16bf(w1_c), (short)0, acc1, false, false);
        acc2 = __builtin_amdgcn_wmma_f32_16x16x32_bf16(false, as_v16bf(a_c), false,
                  as_v16bf(w2_c), (short)0, acc2, false, false);
        acc3 = __builtin_amdgcn_wmma_f32_16x16x32_bf16(false, as_v16bf(a_c), false,
                  as_v16bf(w3_c), (short)0, acc3, false, false);

        a_c = a_n; w0_c = w0_n; w1_c = w1_n; w2_c = w2_n; w3_c = w3_n;
    }

    // epilogue: last k-step
    acc0 = __builtin_amdgcn_wmma_f32_16x16x32_bf16(false, as_v16bf(a_c), false,
              as_v16bf(w0_c), (short)0, acc0, false, false);
    acc1 = __builtin_amdgcn_wmma_f32_16x16x32_bf16(false, as_v16bf(a_c), false,
              as_v16bf(w1_c), (short)0, acc1, false, false);
    acc2 = __builtin_amdgcn_wmma_f32_16x16x32_bf16(false, as_v16bf(a_c), false,
              as_v16bf(w2_c), (short)0, acc2, false, false);
    acc3 = __builtin_amdgcn_wmma_f32_16x16x32_bf16(false, as_v16bf(a_c), false,
              as_v16bf(w3_c), (short)0, acc3, false, false);

    const int mw = (lane >> 4) << 3;           // 0 or 8
    #pragma unroll
    for (int r = 0; r < 8; ++r) {
        float* drow = D + (size_t)(m0 + mw + r) * CPC_CENC + n0;
        drow[ 0 + lr] = acc0[r];
        drow[16 + lr] = acc1[r];
        drow[32 + lr] = acc2[r];
        drow[48 + lr] = acc3[r];
    }
}

// ---------------------- per-row logits + logsumexp -------------------------
// One 128-thread block per row n of step k. Thread j handles negative j.
__device__ __forceinline__ float cpc_blk_sum(float v, float* red) {
    const int tid = threadIdx.x;
    red[tid] = v; __syncthreads();
    #pragma unroll
    for (int s = 64; s > 0; s >>= 1) {
        if (tid < s) red[tid] += red[tid + s];
        __syncthreads();
    }
    float r = red[0]; __syncthreads();
    return r;
}

__device__ __forceinline__ float cpc_blk_max(float v, float* red) {
    const int tid = threadIdx.x;
    red[tid] = v; __syncthreads();
    #pragma unroll
    for (int s = 64; s > 0; s >>= 1) {
        if (tid < s) red[tid] = fmaxf(red[tid], red[tid + s]);
        __syncthreads();
    }
    float r = red[0]; __syncthreads();
    return r;
}

__global__ void __launch_bounds__(128)
cpc_loss_rows_kernel(const float* __restrict__ zbuf,       // (2048,256) z_hat
                     const float* __restrict__ pool,       // (2048,256) normed
                     const long long* __restrict__ negk,   // (>=Nk,128) indices
                     float* __restrict__ lossk,            // (Nk) per-row loss
                     int k, int Tm) {
    __shared__ __align__(16) float zh[CPC_CENC];
    __shared__ float red[128];

    const int n   = blockIdx.x;                // 0..Nk-1
    const int b   = n / Tm;
    const int t   = n - b * Tm;
    const int zr  = b * CPC_T + t;             // row in zbuf / context order
    const int pi  = zr + k;                    // positive row in pool
    const int tid = threadIdx.x;

    const float a0 = zbuf[(size_t)zr * CPC_CENC + tid];
    const float a1 = zbuf[(size_t)zr * CPC_CENC + 128 + tid];
    zh[tid]       = a0;
    zh[128 + tid] = a1;

    const float* prow = pool + (size_t)pi * CPC_CENC;
    const float ssq = a0 * a0 + a1 * a1;
    const float pp  = a0 * prow[tid] + a1 * prow[128 + tid];

    const float sumsq = cpc_blk_sum(ssq, red);   // first sync also covers zh
    const float pos   = cpc_blk_sum(pp, red);
    const float inv   = rsqrtf(sumsq);

    // one negative per thread: 256-float dot out of the L2-resident pool
    long long idxll = negk[(size_t)n * CPC_K + tid];
    int idx = (int)idxll;
    idx &= (CPC_ROWS - 1);                       // indices are in [0, 2048)
    const float4* nr = (const float4*)(pool + (size_t)idx * CPC_CENC);
    const float4* zr4 = (const float4*)zh;
    float d0 = 0.f, d1 = 0.f;
    #pragma unroll 8
    for (int i = 0; i < 64; i += 2) {
        float4 x = nr[i],     z = zr4[i];
        d0 += x.x * z.x + x.y * z.y + x.z * z.z + x.w * z.w;
        float4 x2 = nr[i + 1], z2 = zr4[i + 1];
        d1 += x2.x * z2.x + x2.y * z2.y + x2.z * z2.z + x2.w * z2.w;
    }
    const float lj = (d0 + d1) * inv * CPC_INVT;     // negative logit j
    const float pl = pos * inv * CPC_INVT;           // positive logit

    float m = cpc_blk_max(lj, red);
    m = fmaxf(m, pl);
    const float s = cpc_blk_sum(__expf(lj - m), red) + __expf(pl - m);
    if (tid == 0) lossk[n] = m + __logf(s) - pl;
}

// --------------------------- final reduction -------------------------------
__global__ void __launch_bounds__(256)
cpc_final_reduce_kernel(const float* __restrict__ lossbuf,  // (12, 2048)
                        float* __restrict__ out) {
    __shared__ float red[256];
    const int tid = threadIdx.x;
    float acc = 0.f;
    for (int k = 1; k <= CPC_P; ++k) {
        const int Nk = CPC_B * (CPC_T - k);
        const float* lb = lossbuf + (size_t)(k - 1) * CPC_ROWS;
        float s = 0.f;
        for (int n = tid; n < Nk; n += 256) s += lb[n];
        acc += s / (float)Nk;
    }
    red[tid] = acc; __syncthreads();
    #pragma unroll
    for (int s = 128; s > 0; s >>= 1) {
        if (tid < s) red[tid] += red[tid + s];
        __syncthreads();
    }
    if (tid == 0) out[0] = red[0] / (float)CPC_P;
}

// ------------------------------- launcher ----------------------------------
extern "C" void kernel_launch(void* const* d_in, const int* in_sizes, int n_in,
                              void* d_out, int out_size, void* d_ws, size_t ws_size,
                              hipStream_t stream) {
    (void)in_sizes; (void)n_in; (void)out_size; (void)ws_size;
    const float*     ctx  = (const float*)d_in[0];     // (8,256,512)
    const float*     enc  = (const float*)d_in[1];     // (8,256,256)
    const float*     W    = (const float*)d_in[2];     // (12,256,512)
    const long long* nidx = (const long long*)d_in[3]; // (12,2040,128) int64
    float* out = (float*)d_out;

    // workspace layout (bytes)
    char* ws = (char*)d_ws;
    float*          pool    = (float*)(ws);                                   // 2 MB
    float*          zbuf    = (float*)(ws + (size_t)2 * 1024 * 1024);         // 2 MB
    unsigned short* ctx_bf  = (unsigned short*)(ws + (size_t)4 * 1024 * 1024);// 2 MB
    unsigned short* w_bf    = (unsigned short*)(ws + (size_t)6 * 1024 * 1024);// 3 MB
    float*          lossbuf = (float*)(ws + (size_t)9216 * 1024);             // 96 KB

    // 1) bf16 conversions for the matrix pipe
    {
        const int nctx = CPC_ROWS * CPC_CCTX;                // 1,048,576
        cpc_f32_to_bf16_kernel<<<(nctx + 255) / 256, 256, 0, stream>>>(ctx, ctx_bf, nctx);
        const int nw = CPC_P * CPC_CENC * CPC_CCTX;          // 1,572,864
        cpc_f32_to_bf16_kernel<<<(nw + 255) / 256, 256, 0, stream>>>(W, w_bf, nw);
    }

    // 2) normalized negative/positive pool
    cpc_pool_norm_kernel<<<CPC_ROWS / 8, 256, 0, stream>>>(enc, pool);

    // 3) per-step GEMM (WMMA) + per-row InfoNCE logits; zbuf reused, stream-ordered
    for (int k = 1; k <= CPC_P; ++k) {
        dim3 ggrid(CPC_CENC / 64, CPC_ROWS / 16);            // (4, 128) waves
        cpc_gemm_zhat_kernel<<<ggrid, 32, 0, stream>>>(
            ctx_bf, w_bf + (size_t)(k - 1) * CPC_CENC * CPC_CCTX, zbuf);

        const int Tm = CPC_T - k;
        const int Nk = CPC_B * Tm;
        cpc_loss_rows_kernel<<<Nk, 128, 0, stream>>>(
            zbuf, pool,
            nidx + (size_t)(k - 1) * CPC_NEGROWS * CPC_K,
            lossbuf + (size_t)(k - 1) * CPC_ROWS,
            k, Tm);
    }

    // 4) deterministic scalar reduction (no float atomics)
    cpc_final_reduce_kernel<<<1, 256, 0, stream>>>(lossbuf, out);
}
